// shapeClassifier_51539607784
// MI455X (gfx1250) — compile-verified, matrix operand order checked by reading, not run on hardware
//
#include <hip/hip_runtime.h>

// ---------------------------------------------------------------------------
// Types for CDNA5 WMMA
// ---------------------------------------------------------------------------
typedef __attribute__((ext_vector_type(16))) __bf16 v16bf;
typedef __attribute__((ext_vector_type(8)))  __bf16 v8bf;
typedef __attribute__((ext_vector_type(8)))  float  v8f;
typedef __attribute__((ext_vector_type(4)))  float  v4f;

static __device__ __forceinline__ unsigned short f2bfu(float f) {
    unsigned u = __builtin_bit_cast(unsigned, f);
    unsigned r = (u + 0x7FFFu + ((u >> 16) & 1u)) >> 16;   // round-to-nearest-even
    return (unsigned short)r;
}
static __device__ __forceinline__ __bf16 f2bf(float f) {
    unsigned short s = f2bfu(f);
    return __builtin_bit_cast(__bf16, s);
}

// ---------------------------------------------------------------------------
// Problem constants
// ---------------------------------------------------------------------------
#define NCLOUD   131072      // B*NC = 32*4096
#define NC_      4096
#define KPT      21
#define LPE_     8
#define IN_      231         // (3+8)*21
#define KP0      256         // padded K for layer 0
#define HID_     512
#define OUTC     8
#define OUTP     16

// ---------------------------------------------------------------------------
// Geometry + Laplacian + Jacobi eigensolver + feature assembly.
// One wave32 per point cloud (21 active lanes). 8 clouds per workgroup.
// ---------------------------------------------------------------------------
#define GWAVES 8

__global__ __launch_bounds__(256) void geom_kernel(const float* __restrict__ x,
                                                   unsigned short* __restrict__ feat) {
    __shared__ float sA [GWAVES][KPT][22];
    __shared__ float sV [GWAVES][KPT][22];
    __shared__ float sT [GWAVES][24];
    __shared__ int   sC [GWAVES][24];
    __shared__ float sP [GWAVES][KPT][3];
    __shared__ float sE [GWAVES][KPT][LPE_];

    const int tid  = threadIdx.x;
    const int w    = tid >> 5;
    const int j    = tid & 31;
    const int cloud = blockIdx.x * GWAVES + w;
    const int b = cloud >> 12;          // / NC_
    const int c = cloud & (NC_ - 1);

    float (*A)[22] = sA[w];
    float (*V)[22] = sV[w];

    // ---- load points: x layout (B,3,NC,K) ----
    const size_t base = ((size_t)(b * 3) * NC_ + c) * KPT;
    float px = 0.f, py = 0.f, pz = 0.f, n2 = 0.f;
    if (j < KPT) {
        px = x[base + j];
        py = x[base + (size_t)NC_ * KPT + j];
        pz = x[base + (size_t)2 * NC_ * KPT + j];
        n2 = px * px + py * py + pz * pz;
    }

    // ---- affinity w = exp(dist), column sums ----
    float cs = 0.f;
    for (int i = 0; i < KPT; ++i) {
        float qx = __shfl(px, i, 32), qy = __shfl(py, i, 32);
        float qz = __shfl(pz, i, 32), qn = __shfl(n2, i, 32);
        if (j < KPT) {
            float sq = qn + n2 - 2.f * (qx * px + qy * py + qz * pz);
            sq = fmaxf(sq, 0.f);
            float d  = (sq > 0.f) ? sqrtf(sq) : 0.f;
            float wv = __expf(d);
            A[i][j] = wv;
            cs += wv;
        }
    }
    float inv = (j < KPT) ? rsqrtf(cs + 1e-7f) : 0.f;

    // ---- normalized Laplacian, init V = I ----
    for (int i = 0; i < KPT; ++i) {
        float invi = __shfl(inv, i, 32);
        if (j < KPT) {
            float wv = A[i][j];
            A[i][j] = ((i == j) ? 1.f : 0.f) - invi * wv * inv;
            V[i][j] = (i == j) ? 1.f : 0.f;
        }
    }

    // ---- cyclic Jacobi: A <- J^T A J, V <- V J ----
    for (int sweep = 0; sweep < 10; ++sweep) {
        for (int p = 0; p < KPT - 1; ++p) {
            for (int q = p + 1; q < KPT; ++q) {
                float apq = A[p][q];                   // uniform broadcast read
                if (fabsf(apq) < 1e-12f) continue;     // uniform branch
                float app = A[p][p], aqq = A[q][q];
                float tau = (aqq - app) / (2.f * apq);
                float tt  = ((tau >= 0.f) ? 1.f : -1.f) /
                            (fabsf(tau) + sqrtf(1.f + tau * tau));
                float cc  = rsqrtf(1.f + tt * tt);
                float ssn = tt * cc;
                if (j < KPT) {
                    // row rotation (J^T A): rows p,q
                    float ap = A[p][j], aq = A[q][j];
                    A[p][j] = cc * ap - ssn * aq;
                    A[q][j] = ssn * ap + cc * aq;
                }
                if (j < KPT) {
                    // column rotation (A J): cols p,q (sees updated rows: wave-lockstep LDS)
                    float bp = A[j][p], bq = A[j][q];
                    A[j][p] = cc * bp - ssn * bq;
                    A[j][q] = ssn * bp + cc * bq;
                    float vp = V[j][p], vq = V[j][q];
                    V[j][p] = cc * vp - ssn * vq;
                    V[j][q] = ssn * vp + cc * vq;
                }
            }
        }
    }

    // ---- rank eigenvalues ascending; eigenvector of rank r = column col_of_rank[r] ----
    float lam = 0.f;
    if (j < KPT) { lam = A[j][j]; sT[w][j] = lam; }
    if (j < KPT) {
        int rank = 0;
        for (int i = 0; i < KPT; ++i) {
            float li = sT[w][i];
            rank += (li < lam) || (li == lam && i < j);
        }
        sC[w][rank] = j;
    }
    float evr[LPE_];
    if (j < KPT) {
        for (int m = 0; m < LPE_; ++m) evr[m] = V[j][sC[w][m + 1]];
    }

    // ---- argsort rows 1..20 by |ev0| (row 0 stays first) ----
    float aab = (j < KPT) ? fabsf(evr[0]) : 0.f;
    if (j < KPT) sT[w][j] = aab;
    int pos = 0;
    if (j >= 1 && j < KPT) {
        pos = 1;
        for (int i = 1; i < KPT; ++i) {
            float ai = sT[w][i];
            pos += (ai < aab) || (ai == aab && i < j);
        }
    }
    if (j < KPT) {
        sP[w][pos][0] = px; sP[w][pos][1] = py; sP[w][pos][2] = pz;
        for (int m = 0; m < LPE_; ++m) sE[w][pos][m] = evr[m];
    }

    // ---- assemble features: f[ch*21 + pos], ch 0..2 = xyz, 3..10 = ev; pad to 256, bf16 ----
    unsigned short o8[8];
#pragma unroll
    for (int u = 0; u < 8; ++u) {
        int idx = j * 8 + u;
        float v = 0.f;
        if (idx < IN_) {
            int ch = idx / KPT, ps = idx % KPT;
            v = (ch < 3) ? sP[w][ps][ch] : sE[w][ps][ch - 3];
        }
        o8[u] = f2bfu(v);
    }
    unsigned u0 = (unsigned)o8[0] | ((unsigned)o8[1] << 16);
    unsigned u1 = (unsigned)o8[2] | ((unsigned)o8[3] << 16);
    unsigned u2 = (unsigned)o8[4] | ((unsigned)o8[5] << 16);
    unsigned u3 = (unsigned)o8[6] | ((unsigned)o8[7] << 16);
    uint4 pk = make_uint4(u0, u1, u2, u3);
    *reinterpret_cast<uint4*>(feat + (size_t)cloud * KP0 + j * 8) = pk;
}

// ---------------------------------------------------------------------------
// Weight pack: W (K x Mreal, f32, row-major) -> Wt (Mpad x Kp, bf16), zero padded
// ---------------------------------------------------------------------------
__global__ void pack_weights(const float* __restrict__ W, unsigned short* __restrict__ Wt,
                             int K, int Kp, int Mpad, int Mreal) {
    int idx = blockIdx.x * blockDim.x + threadIdx.x;
    if (idx >= Mpad * Kp) return;
    int m = idx / Kp, k = idx % Kp;
    float v = (k < K && m < Mreal) ? W[(size_t)k * Mreal + m] : 0.f;
    Wt[idx] = f2bfu(v);
}

__global__ void zero_f32(float* __restrict__ p, int n) {
    int i = blockIdx.x * blockDim.x + threadIdx.x;
    if (i < n) p[i] = 0.f;
}

// mean/var -> BN scale/shift:  y = h*s + t ;  s = g*rsqrt(var+eps), t = be - mean*s
__global__ void finalize_bn(const float* __restrict__ sum, const float* __restrict__ ssq,
                            const float* __restrict__ g, const float* __restrict__ be,
                            float* __restrict__ s, float* __restrict__ t, float invN) {
    int i = blockIdx.x * blockDim.x + threadIdx.x;
    if (i >= HID_) return;
    float mean = sum[i] * invN;
    float var  = ssq[i] * invN - mean * mean;
    float rs   = rsqrtf(var + 1e-5f);
    float sv   = g[i] * rs;
    s[i] = sv;
    t[i] = be[i] - mean * sv;
}

// ---------------------------------------------------------------------------
// bf16 WMMA GEMM:  Out[N x Mreal] = act(A) @ W + bias
//   MODE 0: A is bf16 (N x KP), raw          (layer 0)
//   MODE 1: A is f32  (N x KP); per-K BN scale/shift + ReLU fused into bf16 cvt
//   STATS : accumulate per-output-channel sum / sumsq for the next BN
// Wave tile: 16 rows x (COLS*16) cols.  Workgroup: 8 waves = 128 rows.
// KP is compile-time: the two-phase pipelined K-loop fully unrolls, so the
// double-buffer rotation is pure SSA renaming (no v_mov copies, no branches);
// loads for step k+32/k+64 overlap the WMMAs for step k.
// A fragment (ISA 16-bit 16x32): row = lane&15; K halves 0..7 at k+8*(lane>>4),
// halves 8..15 at +16.  B pre-transposed (M x KP) mirrors A with col = lane&15.
// C/D: vgpr r -> row r + 8*(lane>>4), col = lane&15.
// ---------------------------------------------------------------------------
template <int MODE, int COLS, bool STATS, int KP>
__global__ __launch_bounds__(256) void gemm_wmma(
        const void* __restrict__ Ain, const unsigned short* __restrict__ Wt,
        const float* __restrict__ bias,
        const float* __restrict__ bn_s, const float* __restrict__ bn_t,
        float* __restrict__ Out, float* __restrict__ gsum, float* __restrict__ gssq,
        int Mreal) {
    __shared__ float ls[COLS * 16];
    __shared__ float lss[COLS * 16];

    const int tid  = threadIdx.x;
    const int wave = tid >> 5;
    const int lane = tid & 31;
    const int half = lane >> 4;
    const int l15  = lane & 15;
    const int rowTile = blockIdx.x * 128 + wave * 16;
    const int arow    = rowTile + l15;               // A-fragment row for this lane
    const int colbase = blockIdx.y * (COLS * 16);

    if (STATS) {
        if (tid < COLS * 16) { ls[tid] = 0.f; lss[tid] = 0.f; }
        __syncthreads();
    }

    v8f c[COLS];
    const v8f vzero = {0.f, 0.f, 0.f, 0.f, 0.f, 0.f, 0.f, 0.f};
#pragma unroll
    for (int jf = 0; jf < COLS; ++jf) c[jf] = vzero;

    // per-lane base pointers (constant; K offsets fold into load encodings)
    const unsigned short* abp = (const unsigned short*)Ain + (size_t)arow * KP + half * 8;
    const float*          afp = (const float*)Ain          + (size_t)arow * KP + half * 8;
    const float*          sp  = bn_s + half * 8;
    const float*          tp  = bn_t + half * 8;
    const unsigned short* wp[COLS];
#pragma unroll
    for (int jf = 0; jf < COLS; ++jf)
        wp[jf] = Wt + (size_t)(colbase + jf * 16 + l15) * KP + half * 8;

    // ---- fragment loaders ----
    auto loadA = [&](int k) -> v16bf {
        v16bf a;
        if (MODE == 0) {
            const v8bf p0 = *reinterpret_cast<const v8bf*>(abp + k);
            const v8bf p1 = *reinterpret_cast<const v8bf*>(abp + k + 16);
#pragma unroll
            for (int i = 0; i < 8; ++i) { a[i] = p0[i]; a[8 + i] = p1[i]; }
        } else {
            v4f f0 = *reinterpret_cast<const v4f*>(afp + k);
            v4f f1 = *reinterpret_cast<const v4f*>(afp + k + 4);
            v4f f2 = *reinterpret_cast<const v4f*>(afp + k + 16);
            v4f f3 = *reinterpret_cast<const v4f*>(afp + k + 20);
            v4f s0 = *reinterpret_cast<const v4f*>(sp + k);
            v4f s1 = *reinterpret_cast<const v4f*>(sp + k + 4);
            v4f s2 = *reinterpret_cast<const v4f*>(sp + k + 16);
            v4f s3 = *reinterpret_cast<const v4f*>(sp + k + 20);
            v4f t0 = *reinterpret_cast<const v4f*>(tp + k);
            v4f t1 = *reinterpret_cast<const v4f*>(tp + k + 4);
            v4f t2 = *reinterpret_cast<const v4f*>(tp + k + 16);
            v4f t3 = *reinterpret_cast<const v4f*>(tp + k + 20);
#pragma unroll
            for (int i = 0; i < 4; ++i) {
                a[i]      = f2bf(fmaxf(f0[i] * s0[i] + t0[i], 0.f));
                a[4 + i]  = f2bf(fmaxf(f1[i] * s1[i] + t1[i], 0.f));
                a[8 + i]  = f2bf(fmaxf(f2[i] * s2[i] + t2[i], 0.f));
                a[12 + i] = f2bf(fmaxf(f3[i] * s3[i] + t3[i], 0.f));
            }
        }
        return a;
    };
    auto loadB = [&](int jf, int k) -> v16bf {
        const v8bf q0 = *reinterpret_cast<const v8bf*>(wp[jf] + k);
        const v8bf q1 = *reinterpret_cast<const v8bf*>(wp[jf] + k + 16);
        v16bf bf;
#pragma unroll
        for (int i = 0; i < 8; ++i) { bf[i] = q0[i]; bf[8 + i] = q1[i]; }
        return bf;
    };

    // ---- fully-unrolled two-phase software pipeline ----
    v16bf a0 = loadA(0);
    v16bf b0[COLS];
#pragma unroll
    for (int jf = 0; jf < COLS; ++jf) b0[jf] = loadB(jf, 0);

#pragma unroll
    for (int k = 0; k < KP; k += 64) {
        // phase 1: issue loads for k+32, compute on (a0,b0)
        v16bf a1 = loadA(k + 32);
        v16bf b1[COLS];
#pragma unroll
        for (int jf = 0; jf < COLS; ++jf) b1[jf] = loadB(jf, k + 32);
#pragma unroll
        for (int jf = 0; jf < COLS; ++jf)
            c[jf] = __builtin_amdgcn_wmma_f32_16x16x32_bf16(
                        false, a0, false, b0[jf], (short)0, c[jf], false, false);

        // phase 2: issue loads for k+64 (compile-time guarded), compute on (a1,b1)
        if (k + 64 < KP) {
            a0 = loadA(k + 64);
#pragma unroll
            for (int jf = 0; jf < COLS; ++jf) b0[jf] = loadB(jf, k + 64);
        }
#pragma unroll
        for (int jf = 0; jf < COLS; ++jf)
            c[jf] = __builtin_amdgcn_wmma_f32_16x16x32_bf16(
                        false, a1, false, b1[jf], (short)0, c[jf], false, false);
    }

    // ---- bias, store, BN stats ----
#pragma unroll
    for (int jf = 0; jf < COLS; ++jf) {
        const int coln = colbase + jf * 16 + l15;
        const float bv = (coln < Mreal) ? bias[coln] : 0.f;
        float psum = 0.f, pss = 0.f;
#pragma unroll
        for (int r = 0; r < 8; ++r) {
            float v = c[jf][r] + bv;
            psum += v;
            pss  += v * v;
            const int m = r + half * 8;
            if (coln < Mreal)
                Out[(size_t)(rowTile + m) * Mreal + coln] = v;
        }
        if (STATS) {
            atomicAdd(&ls[jf * 16 + l15], psum);
            atomicAdd(&lss[jf * 16 + l15], pss);
        }
    }
    if (STATS) {
        __syncthreads();
        if (tid < COLS * 16) {
            atomicAdd(&gsum[colbase + tid], ls[tid]);
            atomicAdd(&gssq[colbase + tid], lss[tid]);
        }
    }
}

// ---------------------------------------------------------------------------
// Host: workspace layout + launches (all on `stream`, graph-capture safe)
// ---------------------------------------------------------------------------
extern "C" void kernel_launch(void* const* d_in, const int* in_sizes, int n_in,
                              void* d_out, int out_size, void* d_ws, size_t ws_size,
                              hipStream_t stream) {
    (void)in_sizes; (void)n_in; (void)out_size; (void)ws_size;

    const float* x   = (const float*)d_in[0];
    const float* W0  = (const float*)d_in[1];
    const float* b0  = (const float*)d_in[2];
    const float* g0  = (const float*)d_in[3];
    const float* be0 = (const float*)d_in[4];
    const float* W1  = (const float*)d_in[5];
    const float* b1  = (const float*)d_in[6];
    const float* g1  = (const float*)d_in[7];
    const float* be1 = (const float*)d_in[8];
    const float* W2  = (const float*)d_in[9];
    const float* b2  = (const float*)d_in[10];
    const float* g2  = (const float*)d_in[11];
    const float* be2 = (const float*)d_in[12];
    const float* W3  = (const float*)d_in[13];
    const float* b3  = (const float*)d_in[14];

    unsigned char* ws = (unsigned char*)d_ws;
    size_t off = 0;
    unsigned short* feat = (unsigned short*)(ws + off); off += (size_t)NCLOUD * KP0 * 2;
    float* h0            = (float*)(ws + off);          off += (size_t)NCLOUD * HID_ * 4;
    float* h1            = (float*)(ws + off);          off += (size_t)NCLOUD * HID_ * 4;
    unsigned short* wt0  = (unsigned short*)(ws + off); off += (size_t)HID_ * KP0 * 2;
    unsigned short* wt1  = (unsigned short*)(ws + off); off += (size_t)HID_ * HID_ * 2;
    unsigned short* wt2  = (unsigned short*)(ws + off); off += (size_t)HID_ * HID_ * 2;
    unsigned short* wt3  = (unsigned short*)(ws + off); off += (size_t)OUTP * HID_ * 2;
    float* stats         = (float*)(ws + off);          off += (size_t)3 * 4 * HID_ * 4;

    float* sum0 = stats + 0 * 2048, *ssq0 = sum0 + 512, *s0 = sum0 + 1024, *t0 = sum0 + 1536;
    float* sum1 = stats + 1 * 2048, *ssq1 = sum1 + 512, *s1 = sum1 + 1024, *t1 = sum1 + 1536;
    float* sum2 = stats + 2 * 2048, *ssq2 = sum2 + 512, *s2 = sum2 + 1024, *t2 = sum2 + 1536;

    const float invN = 1.f / (float)NCLOUD;

    // 0) zero stats accumulators (deterministic each call)
    zero_f32<<<(3 * 2048 + 255) / 256, 256, 0, stream>>>(stats, 3 * 2048);

    // 1) pack weights to transposed bf16 (M x Kp)
    pack_weights<<<(HID_ * KP0  + 255) / 256, 256, 0, stream>>>(W0, wt0, IN_,  KP0,  HID_, HID_);
    pack_weights<<<(HID_ * HID_ + 255) / 256, 256, 0, stream>>>(W1, wt1, HID_, HID_, HID_, HID_);
    pack_weights<<<(HID_ * HID_ + 255) / 256, 256, 0, stream>>>(W2, wt2, HID_, HID_, HID_, HID_);
    pack_weights<<<(OUTP * HID_ + 255) / 256, 256, 0, stream>>>(W3, wt3, HID_, HID_, OUTP, OUTC);

    // 2) geometry + spectral features (one wave32 per cloud)
    geom_kernel<<<NCLOUD / GWAVES, 256, 0, stream>>>(x, feat);

    // 3) layer 0: feat(bf16, KP=256) @ W0 + b0 -> h0 ; stats0
    gemm_wmma<0, 4, true, KP0><<<dim3(NCLOUD / 128, HID_ / 64), 256, 0, stream>>>(
        feat, wt0, b0, nullptr, nullptr, h0, sum0, ssq0, HID_);
    finalize_bn<<<2, 256, 0, stream>>>(sum0, ssq0, g0, be0, s0, t0, invN);

    // 4) layer 1: relu(bn(h0)) @ W1 + b1 -> h1 ; stats1
    gemm_wmma<1, 4, true, HID_><<<dim3(NCLOUD / 128, HID_ / 64), 256, 0, stream>>>(
        h0, wt1, b1, s0, t0, h1, sum1, ssq1, HID_);
    finalize_bn<<<2, 256, 0, stream>>>(sum1, ssq1, g1, be1, s1, t1, invN);

    // 5) layer 2: relu(bn(h1)) @ W2 + b2 -> h0 (reuse) ; stats2
    gemm_wmma<1, 4, true, HID_><<<dim3(NCLOUD / 128, HID_ / 64), 256, 0, stream>>>(
        h1, wt2, b2, s1, t1, h0, sum2, ssq2, HID_);
    finalize_bn<<<2, 256, 0, stream>>>(sum2, ssq2, g2, be2, s2, t2, invN);

    // 6) layer 3: relu(bn(h0)) @ W3 + b3 -> d_out (N x 8)
    gemm_wmma<1, 1, false, HID_><<<dim3(NCLOUD / 128, 1), 256, 0, stream>>>(
        h0, wt3, b3, s2, t2, (float*)d_out, nullptr, nullptr, OUTC);
}
